// DualLSTMTagger_16484084482265
// MI455X (gfx1250) — matrix operand order repaired and checked by reading, not compiled, not run on hardware
//
#include <hip/hip_runtime.h>

typedef __attribute__((ext_vector_type(16))) __bf16 v16bf;
typedef __attribute__((ext_vector_type(8)))  float  v8f;
typedef __attribute__((ext_vector_type(4)))  unsigned int v4u;
typedef __attribute__((ext_vector_type(8)))  int v8i_;
typedef __attribute__((ext_vector_type(4)))  int v4i_;

#define S_WORDS 2048
#define C_CHARS 16
#define DW 256
#define HW 512
#define DC 64
#define HC 128
#define NTAG 50
#define GW (4*HW)   /* 2048 */
#define GC (4*HC)   /* 512  */
#define KW (DW+HC)  /* 384  */

// -------- workspace layout (bytes, all 16B aligned) --------
#define WIHC_OFF  0u          /* [512 x 64]  bf16 */
#define WHHC_OFF  65536u      /* [512 x 128] bf16 */
#define WIHW_OFF  196608u     /* [2048 x 384] bf16 */
#define WHHW_OFF  1769472u    /* [2048 x 512] bf16 */
#define WTAG_OFF  3866624u    /* [64 x 512] bf16 (rows 50..63 zero) */
#define COMB_OFF  3932160u    /* [2048 x 384] bf16 */
#define HWOUT_OFF 5505024u    /* [2048 x 512] bf16 */
#define GX_OFF    7602176u    /* [2048 x 2048] f32 */

// ---------------- helpers ----------------
__device__ __forceinline__ float sigf(float x)   { return 1.f / (1.f + __expf(-x)); }
__device__ __forceinline__ float tanh_f(float x) { return 1.f - 2.f / (__expf(2.f * x) + 1.f); }

union FragU { uint4 q[2]; v16bf v; };

// A fragment 16x32 bf16, row-major src with leading dim ld (elements)
__device__ __forceinline__ v16bf load_a16x32(const __bf16* src, int ld, int m0, int k0) {
  int lane = threadIdx.x & 31;
  int m  = m0 + (lane & 15);
  int kb = k0 + ((lane >> 4) << 3);          // lanes>=16 start at K+8
  FragU u;
  u.q[0] = *reinterpret_cast<const uint4*>(src + m * ld + kb);       // K kb..kb+7
  u.q[1] = *reinterpret_cast<const uint4*>(src + m * ld + kb + 16);  // K kb+16..kb+23
  return u.v;
}

// B fragment 32x16 bf16 from weight stored [N_total x K_total] row-major (B = W^T)
__device__ __forceinline__ v16bf load_b32x16(const __bf16* w, int ldk, int n0, int k0) {
  int lane = threadIdx.x & 31;
  int n  = n0 + (lane & 15);
  int kb = k0 + ((lane >> 4) << 3);          // lanes>=16 start at K+8
  FragU u;
  const uint4* p = reinterpret_cast<const uint4*>(w + n * ldk + kb);
  u.q[0] = p[0];
  u.q[1] = p[2];
  return u.v;
}

__device__ __forceinline__ v8f wmma_bf16(v16bf a, v16bf b, v8f c) {
  return __builtin_amdgcn_wmma_f32_16x16x32_bf16(false, a, false, b, (short)0, c, false, false);
}

// ---------------- Tensor Data Mover: 1-D bf16 tile -> LDS ----------------
#if __has_builtin(__builtin_amdgcn_tensor_load_to_lds)
#define HAVE_TDM 1
// Copies nelem bf16 elements (nelem < 65536) from contiguous global memory to LDS.
// D# per CDNA5 ISA 8.3/8.4: group0 = {flags, lds_addr, gaddr_lo, gaddr_hi|type=2},
// group1 packs data_size=2B, tensor_dim0=tile_dim0=nelem, tensor_dim1=tile_dim1=1,
// tensor_dim0_stride=nelem. Groups 2/3 zero (<=2D tensor).
__device__ __forceinline__ void tdm_load_linear_bf16(unsigned lds_byte_addr,
                                                     const void* gsrc,
                                                     unsigned nelem) {
  unsigned long long ga = (unsigned long long)(uintptr_t)gsrc;
  v4u g0;
  g0[0] = 1u;                                                  // count=1, user mode
  g0[1] = lds_byte_addr;                                       // lds_addr
  g0[2] = (unsigned)ga;                                        // global_addr[31:0]
  g0[3] = (unsigned)((ga >> 32) & 0x01ffffffu) | (2u << 30);   // addr[56:32], type=2
  v8i_ g1;
  g1[0] = (int)(1u << 16);                                     // data_size=1 (2 bytes), wg_mask=0
  g1[1] = (int)((nelem & 0xffffu) << 16);                      // tensor_dim0[15:0] -> bits 63:48
  g1[2] = (int)((nelem >> 16) | (1u << 16));                   // tensor_dim0[31:16], tensor_dim1[15:0]=1
  g1[3] = (int)((nelem & 0xffffu) << 16);                      // tensor_dim1[31:16]=0, tile_dim0=nelem
  g1[4] = 1;                                                   // tile_dim1=1, tile_dim2=0
  g1[5] = (int)nelem;                                          // tensor_dim0_stride[31:0]
  g1[6] = 0;                                                   // stride0[47:32]=0, stride1[15:0]=0
  g1[7] = 0;
  v4i_ z4 = {0, 0, 0, 0};
#if defined(__clang_major__) && (__clang_major__ >= 23)
  v8i_ z8 = {0, 0, 0, 0, 0, 0, 0, 0};
  __builtin_amdgcn_tensor_load_to_lds(g0, g1, z4, z4, z8, 0);
#else
  __builtin_amdgcn_tensor_load_to_lds(g0, g1, z4, z4, 0);
#endif
}
#endif

// ---------------- prep kernels ----------------
__global__ void cvt_f32_bf16(__bf16* dst, const float* src, int n) {
  int i = blockIdx.x * blockDim.x + threadIdx.x;
  if (i < n) dst[i] = (__bf16)src[i];
}

__global__ void wtag_prep(__bf16* dst, const float* wtag) {  // 64x512 zero-padded
  int i = blockIdx.x * blockDim.x + threadIdx.x;
  if (i < 64 * HW) {
    int r = i >> 9, c = i & 511;
    dst[i] = (r < NTAG) ? (__bf16)wtag[r * HW + c] : (__bf16)0.0f;
  }
}

__global__ void word_embed(__bf16* comb, const float* wemb, const int* sent) {
  int i = blockIdx.x * blockDim.x + threadIdx.x;
  if (i < S_WORDS * DW) {
    int s = i >> 8, d = i & 255;
    comb[s * KW + d] = (__bf16)wemb[sent[s] * DW + d];
  }
}

// ---------------- char LSTM: 16 words per workgroup, 8 waves x 64 gate cols ----------------
__global__ __launch_bounds__(256) void char_lstm(
    const int* __restrict__ char_ids, const int* __restrict__ wlen,
    const float* __restrict__ cemb,
    const __bf16* __restrict__ Wihc, const __bf16* __restrict__ Whhc,
    const float* __restrict__ bc, __bf16* __restrict__ comb)
{
  __shared__ float gsm[16][GC];                            // 32 KB gates f32
  __shared__ float csm[16][HC];                            // 8 KB cell state
  __shared__ __align__(16) __bf16 hsm[16][HC + 8];         // h, ld 136
  __shared__ __align__(16) __bf16 xsm[16][DC + 8];         // x_t, ld 72
  __shared__ int ids[16][C_CHARS];
  __shared__ int lens[16];

  const int tid = threadIdx.x;
  const int lane = tid & 31;
  const int w = tid >> 5;
  const int s0 = blockIdx.x * 16;

  for (int e = tid; e < 16 * C_CHARS; e += 256)
    ids[e >> 4][e & 15] = char_ids[(s0 + (e >> 4)) * C_CHARS + (e & 15)];
  if (tid < 16) lens[tid] = wlen[s0 + tid];
  for (int e = tid; e < 16 * HC; e += 256) {
    int m = e >> 7, j = e & 127;
    csm[m][j] = 0.f;
    hsm[m][j] = (__bf16)0.0f;
  }
  __syncthreads();

  for (int t = 0; t < C_CHARS; ++t) {
    // gather char embeddings for this timestep
    for (int e = tid; e < 16 * DC; e += 256) {
      int m = e >> 6, k = e & 63;
      xsm[m][k] = (__bf16)cemb[ids[m][t] * DC + k];
    }
    __syncthreads();

    v8f acc[4];
    #pragma unroll
    for (int j = 0; j < 4; ++j)
      #pragma unroll
      for (int r = 0; r < 8; ++r) acc[j][r] = 0.f;

    #pragma unroll
    for (int kc = 0; kc < 2; ++kc) {                       // x part, K=64
      v16bf a = load_a16x32(&xsm[0][0], DC + 8, 0, kc * 32);
      #pragma unroll
      for (int j = 0; j < 4; ++j) {
        v16bf b = load_b32x16(Wihc, DC, w * 64 + j * 16, kc * 32);
        acc[j] = wmma_bf16(a, b, acc[j]);
      }
    }
    #pragma unroll
    for (int kc = 0; kc < 4; ++kc) {                       // h part, K=128
      v16bf a = load_a16x32(&hsm[0][0], HC + 8, 0, kc * 32);
      #pragma unroll
      for (int j = 0; j < 4; ++j) {
        v16bf b = load_b32x16(Whhc, HC, w * 64 + j * 16, kc * 32);
        acc[j] = wmma_bf16(a, b, acc[j]);
      }
    }
    {
      int mo = (lane < 16) ? 0 : 8, nn = lane & 15;
      #pragma unroll
      for (int j = 0; j < 4; ++j)
        #pragma unroll
        for (int r = 0; r < 8; ++r)
          gsm[r + mo][w * 64 + j * 16 + nn] = acc[j][r];
    }
    __syncthreads();

    // LSTM cell with per-word length mask
    for (int e = tid; e < 16 * HC; e += 256) {
      int m = e >> 7, j = e & 127;
      if (t < lens[m]) {
        float ig = sigf(gsm[m][j]            + bc[j]);
        float fg = sigf(gsm[m][HC + j]       + bc[HC + j]);
        float gg = tanh_f(gsm[m][2 * HC + j] + bc[2 * HC + j]);
        float og = sigf(gsm[m][3 * HC + j]   + bc[3 * HC + j]);
        float c = fg * csm[m][j] + ig * gg;
        csm[m][j] = c;
        hsm[m][j] = (__bf16)(og * tanh_f(c));
      }
    }
    __syncthreads();
  }

  // final h -> combined[:, DW:DW+HC]
  for (int e = tid; e < 16 * HC; e += 256) {
    int m = e >> 7, j = e & 127;
    comb[(s0 + m) * KW + DW + j] = hsm[m][j];
  }
}

// ---------------- Gx = combined @ W_ih_w^T + b_w  (2048x2048, K=384) ----------------
// A tile (64x384) and B tile (64x384) are DMA'd to LDS by the Tensor Data Mover,
// then all WMMA operand traffic is ds_load_b128.
__global__ __launch_bounds__(256) void gx_gemm(
    const __bf16* __restrict__ comb, const __bf16* __restrict__ Wihw,
    const float* __restrict__ bw, float* __restrict__ Gx)
{
  const int tid = threadIdx.x, lane = tid & 31, w = tid >> 5;
  const int m0 = blockIdx.y * 64 + (w >> 1) * 16;
  const int n0 = blockIdx.x * 64 + (w & 1) * 32;

#ifdef HAVE_TDM
  __shared__ __align__(16) __bf16 a_sm[64 * KW];   // 48 KB
  __shared__ __align__(16) __bf16 b_sm[64 * KW];   // 48 KB
  if (w == 0) {
    tdm_load_linear_bf16((unsigned)(uintptr_t)(void*)a_sm,
                         comb + (size_t)blockIdx.y * 64 * KW, 64 * KW);
    tdm_load_linear_bf16((unsigned)(uintptr_t)(void*)b_sm,
                         Wihw + (size_t)blockIdx.x * 64 * KW, 64 * KW);
    __builtin_amdgcn_s_wait_tensorcnt(0);
  }
  __syncthreads();
  const __bf16* Abase = a_sm;
  const __bf16* Bbase = b_sm;
  const int am0 = (w >> 1) * 16;
  const int bn0 = (w & 1) * 32;
#else
  const __bf16* Abase = comb;
  const __bf16* Bbase = Wihw;
  const int am0 = m0;
  const int bn0 = n0;
#endif

  v8f acc0, acc1;
  #pragma unroll
  for (int r = 0; r < 8; ++r) { acc0[r] = 0.f; acc1[r] = 0.f; }
  #pragma unroll 4
  for (int kc = 0; kc < KW / 32; ++kc) {
    v16bf a  = load_a16x32(Abase, KW, am0, kc * 32);
    v16bf b0 = load_b32x16(Bbase, KW, bn0,      kc * 32);
    v16bf b1 = load_b32x16(Bbase, KW, bn0 + 16, kc * 32);
    acc0 = wmma_bf16(a, b0, acc0);
    acc1 = wmma_bf16(a, b1, acc1);
  }
  int mo = (lane < 16) ? 0 : 8, nn = lane & 15;
  #pragma unroll
  for (int r = 0; r < 8; ++r) {
    int row = m0 + mo + r;
    Gx[row * GW + n0 + nn]      = acc0[r] + bw[n0 + nn];
    Gx[row * GW + n0 + 16 + nn] = acc1[r] + bw[n0 + 16 + nn];
  }
}

// ---------------- word LSTM recurrence: 1 persistent WG, 32 waves x 64 gate cols ----------------
__global__ __launch_bounds__(1024) void word_lstm(
    const float* __restrict__ Gx, const __bf16* __restrict__ Whhw,
    __bf16* __restrict__ Hw)
{
  __shared__ __align__(16) __bf16 hsm[HW];
  __shared__ float csm[HW];
  __shared__ float gsm[GW];
  const int tid = threadIdx.x, lane = tid & 31, w = tid >> 5;
  if (tid < HW) { hsm[tid] = (__bf16)0.0f; csm[tid] = 0.f; }
  __syncthreads();
  const int n0 = w * 64;

  for (int t = 0; t < S_WORDS; ++t) {
    // prefetch next step's Gx row (8 KB = 64 cachelines) while this step computes
    if (t + 1 < S_WORDS && tid < 64)
      __builtin_prefetch(Gx + (size_t)(t + 1) * GW + tid * 32, 0, 1);

    v8f acc[4];
    #pragma unroll
    for (int j = 0; j < 4; ++j)
      #pragma unroll
      for (int r = 0; r < 8; ++r) acc[j][r] = 0.f;

    for (int kc = 0; kc < HW / 32; ++kc) {
      // A: only row 0 valid (h vector); rows 1..15 zero
      v16bf a;
      #pragma unroll
      for (int i = 0; i < 16; ++i) a[i] = (__bf16)0.0f;
      if ((lane & 15) == 0) {
        int kb = kc * 32 + ((lane >> 4) << 3);
        #pragma unroll
        for (int i = 0; i < 8; ++i) a[i]     = hsm[kb + i];
        #pragma unroll
        for (int i = 0; i < 8; ++i) a[8 + i] = hsm[kb + 16 + i];
      }
      #pragma unroll
      for (int j = 0; j < 4; ++j) {
        v16bf b = load_b32x16(Whhw, HW, n0 + j * 16, kc * 32);
        acc[j] = wmma_bf16(a, b, acc[j]);
      }
    }
    if (lane < 16) {                 // D row 0 lives in VGPR0, lanes 0..15
      #pragma unroll
      for (int j = 0; j < 4; ++j) {
        int n = n0 + j * 16 + lane;
        gsm[n] = acc[j][0] + Gx[(size_t)t * GW + n];   // b_w already folded into Gx
      }
    }
    __syncthreads();
    if (tid < HW) {
      int j = tid;
      float ig = sigf(gsm[j]);
      float fg = sigf(gsm[HW + j]);
      float gg = tanh_f(gsm[2 * HW + j]);
      float og = sigf(gsm[3 * HW + j]);
      float c = fg * csm[j] + ig * gg;
      csm[j] = c;
      __bf16 hb = (__bf16)(og * tanh_f(c));
      hsm[j] = hb;
      Hw[(size_t)t * HW + j] = hb;
    }
    __syncthreads();
  }
}

// ---------------- tag projection + log-softmax ----------------
__global__ __launch_bounds__(256) void tag_softmax(
    const __bf16* __restrict__ Hw, const __bf16* __restrict__ Wtag,
    const float* __restrict__ btag, float* __restrict__ out)
{
  __shared__ float lsm[16][64];
  const int tid = threadIdx.x, lane = tid & 31, w = tid >> 5;
  const int s0 = blockIdx.x * 16;

  if (w < 4) {                         // waves 0..3 each own one 16-col tile
    v8f acc;
    #pragma unroll
    for (int r = 0; r < 8; ++r) acc[r] = 0.f;
    for (int kc = 0; kc < HW / 32; ++kc) {
      v16bf a = load_a16x32(Hw, HW, s0, kc * 32);
      v16bf b = load_b32x16(Wtag, HW, w * 16, kc * 32);
      acc = wmma_bf16(a, b, acc);
    }
    int mo = (lane < 16) ? 0 : 8, nn = lane & 15;
    int col = w * 16 + nn;
    float bt = (col < NTAG) ? btag[col] : 0.f;
    #pragma unroll
    for (int r = 0; r < 8; ++r) lsm[mo + r][col] = acc[r] + bt;
  }
  __syncthreads();

  if (tid < 16) {
    int m = tid;
    float mx = -1e30f;
    for (int j = 0; j < NTAG; ++j) mx = fmaxf(mx, lsm[m][j]);
    float sum = 0.f;
    for (int j = 0; j < NTAG; ++j) sum += __expf(lsm[m][j] - mx);
    float lse = mx + __logf(sum);
    for (int j = 0; j < NTAG; ++j) out[(s0 + m) * NTAG + j] = lsm[m][j] - lse;
  }
}

// ---------------- host launch ----------------
extern "C" void kernel_launch(void* const* d_in, const int* in_sizes, int n_in,
                              void* d_out, int out_size, void* d_ws, size_t ws_size,
                              hipStream_t stream)
{
  const int*   sentence = (const int*)  d_in[0];
  const int*   char_ids = (const int*)  d_in[1];
  const int*   wlen     = (const int*)  d_in[2];
  const float* wemb     = (const float*)d_in[3];
  const float* cemb     = (const float*)d_in[4];
  const float* Wihc     = (const float*)d_in[5];
  const float* Whhc     = (const float*)d_in[6];
  const float* bc       = (const float*)d_in[7];
  const float* Wihw     = (const float*)d_in[8];
  const float* Whhw     = (const float*)d_in[9];
  const float* bw       = (const float*)d_in[10];
  const float* Wtag     = (const float*)d_in[11];
  const float* btag     = (const float*)d_in[12];
  float* out = (float*)d_out;
  char*  ws  = (char*)d_ws;

  __bf16* wihc_bf = (__bf16*)(ws + WIHC_OFF);
  __bf16* whhc_bf = (__bf16*)(ws + WHHC_OFF);
  __bf16* wihw_bf = (__bf16*)(ws + WIHW_OFF);
  __bf16* whhw_bf = (__bf16*)(ws + WHHW_OFF);
  __bf16* wtag_bf = (__bf16*)(ws + WTAG_OFF);
  __bf16* comb_bf = (__bf16*)(ws + COMB_OFF);
  __bf16* hw_bf   = (__bf16*)(ws + HWOUT_OFF);
  float*  gx      = (float*)(ws + GX_OFF);

  // weight conversions to bf16
  cvt_f32_bf16<<<(GC * DC + 255) / 256, 256, 0, stream>>>(wihc_bf, Wihc, GC * DC);
  cvt_f32_bf16<<<(GC * HC + 255) / 256, 256, 0, stream>>>(whhc_bf, Whhc, GC * HC);
  cvt_f32_bf16<<<(GW * KW + 255) / 256, 256, 0, stream>>>(wihw_bf, Wihw, GW * KW);
  cvt_f32_bf16<<<(GW * HW + 255) / 256, 256, 0, stream>>>(whhw_bf, Whhw, GW * HW);
  wtag_prep   <<<(64 * HW + 255) / 256, 256, 0, stream>>>(wtag_bf, Wtag);

  // word embeddings into combined[:, 0:256]
  word_embed<<<(S_WORDS * DW + 255) / 256, 256, 0, stream>>>(comb_bf, wemb, sentence);

  // char LSTM -> combined[:, 256:384]
  char_lstm<<<S_WORDS / 16, 256, 0, stream>>>(char_ids, wlen, cemb,
                                              wihc_bf, whhc_bf, bc, comb_bf);

  // Gx = combined @ W_ih_w^T + b_w (TDM-staged tiles)
  gx_gemm<<<dim3(GW / 64, S_WORDS / 64), 256, 0, stream>>>(comb_bf, wihw_bf, bw, gx);

  // sequential word LSTM recurrence
  word_lstm<<<1, 1024, 0, stream>>>(gx, whhw_bf, hw_bf);

  // tag projection + log-softmax
  tag_softmax<<<S_WORDS / 16, 256, 0, stream>>>(hw_bf, wtag_bf, btag, out);
}
